// SGL_22909355557015
// MI455X (gfx1250) — matrix operand order repaired and checked by reading, not compiled, not run on hardware
//
#include <hip/hip_runtime.h>
#include <hip/hip_bf16.h>

// ---------------------------------------------------------------------------
// Types for CDNA5 WMMA
// ---------------------------------------------------------------------------
typedef __attribute__((ext_vector_type(16))) __bf16 v16bf;
typedef __attribute__((ext_vector_type(8)))  float  v8f;

union AF {            // one bf16 WMMA operand fragment (32 bytes / lane)
    v16bf v;
    uint4 u[2];
};

// Manual f32 <-> bf16 (round-to-nearest-even) so we don't depend on __bf16
// arithmetic conversions; the split (hi, lo) gives ~2^-16 relative error.
static __device__ __forceinline__ unsigned short f32_to_bf16_rn(float f) {
    unsigned int u = __float_as_uint(f);
    unsigned int r = u + 0x7FFFu + ((u >> 16) & 1u);
    return (unsigned short)(r >> 16);
}
static __device__ __forceinline__ float bf16_to_f32(unsigned short h) {
    return __uint_as_float(((unsigned int)h) << 16);
}

// ---------------------------------------------------------------------------
// 1) Convert x [N,128] f32 -> split bf16 (hi, lo), float4-vectorized
// ---------------------------------------------------------------------------
__global__ void k_convert_x(const float* __restrict__ x,
                            unsigned short* __restrict__ xhi,
                            unsigned short* __restrict__ xlo,
                            int n4) {
    int i = blockIdx.x * blockDim.x + threadIdx.x;
    if (i >= n4) return;
    float4 v = ((const float4*)x)[i];
    ushort4 h, l;
    h.x = f32_to_bf16_rn(v.x); l.x = f32_to_bf16_rn(v.x - bf16_to_f32(h.x));
    h.y = f32_to_bf16_rn(v.y); l.y = f32_to_bf16_rn(v.y - bf16_to_f32(h.y));
    h.z = f32_to_bf16_rn(v.z); l.z = f32_to_bf16_rn(v.z - bf16_to_f32(h.z));
    h.w = f32_to_bf16_rn(v.w); l.w = f32_to_bf16_rn(v.w - bf16_to_f32(h.w));
    ((ushort4*)xhi)[i] = h;
    ((ushort4*)xlo)[i] = l;
}

// ---------------------------------------------------------------------------
// 2) Pre-swizzle combined B = [wA | wB] (128x128) into fragment-ready layout:
//    frag index ((t*4+kt)*32+lane)*16 + j holds B[kt*32+(j>>3)*16+(lane>>4)*8+(j&7)][t*16+(lane&15)]
//    so each lane's 16 bf16 B values are one contiguous 32B chunk.
// ---------------------------------------------------------------------------
__global__ void k_prep_B(const float* __restrict__ wA,   // [128,64]
                         const float* __restrict__ wB,   // [128,64]
                         unsigned short* __restrict__ Bhi,
                         unsigned short* __restrict__ Blo) {
    int idx = blockIdx.x * blockDim.x + threadIdx.x;     // 0..16383
    if (idx >= 8 * 4 * 32 * 16) return;
    int j    = idx & 15;
    int lane = (idx >> 4) & 31;
    int kt   = (idx >> 9) & 3;
    int t    = (idx >> 11) & 7;
    int k = kt * 32 + (j >> 3) * 16 + (lane >> 4) * 8 + (j & 7); // 0..127
    int n = t * 16 + (lane & 15);                        // 0..127
    float v = (n < 64) ? wA[k * 64 + n] : wB[k * 64 + (n - 64)];
    unsigned short h = f32_to_bf16_rn(v);
    Bhi[idx] = h;
    Blo[idx] = f32_to_bf16_rn(v - bf16_to_f32(h));
}

// ---------------------------------------------------------------------------
// 3) Graph normalization: deg, dinv, norm
// ---------------------------------------------------------------------------
__global__ void k_deg(const int* __restrict__ col, float* __restrict__ deg, int E) {
    int e = blockIdx.x * blockDim.x + threadIdx.x;
    if (e < E) atomicAdd(&deg[col[e]], 1.0f);
}
__global__ void k_dinv(const float* __restrict__ deg, float* __restrict__ dinv, int N) {
    int i = blockIdx.x * blockDim.x + threadIdx.x;
    if (i < N) {
        float d = deg[i];
        dinv[i] = (d > 0.f) ? rsqrtf(fmaxf(d, 1.0f)) : 0.f;
    }
}
__global__ void k_norm(const int* __restrict__ row, const int* __restrict__ col,
                       const float* __restrict__ dinv, float* __restrict__ norm, int E) {
    int e = blockIdx.x * blockDim.x + threadIdx.x;
    if (e < E) norm[e] = dinv[row[e]] * dinv[col[e]];
}

// ---------------------------------------------------------------------------
// 4) Fused dual GEMM: [N,128] x [128,128] where B = [fc_w | conv_w].
//    Split-bf16: C = Ahi*Bhi + Ahi*Blo + Alo*Bhi (f32 accumulate).
//    One wave computes a 16-row x 128-col strip (8 N-tiles, 4 K-tiles).
//    MODE 0: cols 0-63 -> relu(+bias) -> bf16 hi/lo into x3[:, :64]; cols 64-127 -> h (f32)
//    MODE 1: cols 0-63 -> relu(+bias) -> f32 x5;                      cols 64-127 -> h (f32)
// ---------------------------------------------------------------------------
template <int MODE>
__global__ void k_gemm(const unsigned short* __restrict__ Ahi,
                       const unsigned short* __restrict__ Alo,
                       const unsigned short* __restrict__ Bhi,
                       const unsigned short* __restrict__ Blo,
                       const float* __restrict__ bias,        // [64]
                       unsigned short* __restrict__ o_hi,     // MODE 0: [N,128] bf16 hi
                       unsigned short* __restrict__ o_lo,     // MODE 0: [N,128] bf16 lo
                       float* __restrict__ o_relu,            // MODE 1: [N,64]
                       float* __restrict__ o_raw,             // [N,64] pre-scatter h
                       int nstrips) {
    int wave = (int)((blockIdx.x * blockDim.x + threadIdx.x) >> 5);
    int lane = threadIdx.x & 31;
    if (wave >= nstrips) return;                // whole-wave uniform exit
    int half = lane >> 4;
    int mlo  = lane & 15;
    size_t m = (size_t)wave * 16 + mlo;

    // A fragments: ISA 16-bit A 16x32 layout == two contiguous 16B chunks of
    // the row per K-tile: dwords [kt*16+half*4, +4) and [kt*16+8+half*4, +4).
    const uint32_t* rowh = (const uint32_t*)(Ahi + m * 128);
    const uint32_t* rowl = (const uint32_t*)(Alo + m * 128);
    AF ah[4], al[4];
#pragma unroll
    for (int kt = 0; kt < 4; ++kt) {
        int off = kt * 16 + half * 4;
        ah[kt].u[0] = *(const uint4*)(rowh + off);
        ah[kt].u[1] = *(const uint4*)(rowh + off + 8);
        al[kt].u[0] = *(const uint4*)(rowl + off);
        al[kt].u[1] = *(const uint4*)(rowl + off + 8);
    }

#pragma unroll
    for (int t = 0; t < 8; ++t) {
        v8f c = {};
#pragma unroll
        for (int kt = 0; kt < 4; ++kt) {
            AF bh, bl;
            const uint4* pb = (const uint4*)(Bhi + (size_t)((t * 4 + kt) * 32 + lane) * 16);
            const uint4* pl = (const uint4*)(Blo + (size_t)((t * 4 + kt) * 32 + lane) * 16);
            bh.u[0] = pb[0]; bh.u[1] = pb[1];
            bl.u[0] = pl[0]; bl.u[1] = pl[1];
            c = __builtin_amdgcn_wmma_f32_16x16x32_bf16(false, ah[kt].v, false, bh.v,
                                                        (short)0, c, false, false);
            c = __builtin_amdgcn_wmma_f32_16x16x32_bf16(false, ah[kt].v, false, bl.v,
                                                        (short)0, c, false, false);
            c = __builtin_amdgcn_wmma_f32_16x16x32_bf16(false, al[kt].v, false, bh.v,
                                                        (short)0, c, false, false);
        }
        // Epilogue. D layout: VGPR r -> row (r + half*8), col = t*16 + mlo.
        int n_local = t * 16 + mlo;             // 0..127
#pragma unroll
        for (int r = 0; r < 8; ++r) {
            size_t rowo = (size_t)wave * 16 + r + half * 8;
            float v = c[r];
            if (t < 4) {
                v += bias[n_local];
                v = v > 0.f ? v : 0.f;
                if (MODE == 0) {
                    unsigned short h = f32_to_bf16_rn(v);
                    o_hi[rowo * 128 + n_local] = h;
                    o_lo[rowo * 128 + n_local] = f32_to_bf16_rn(v - bf16_to_f32(h));
                } else {
                    o_relu[rowo * 64 + n_local] = v;
                }
            } else {
                o_raw[rowo * 64 + (n_local - 64)] = v;
            }
        }
    }
}

// ---------------------------------------------------------------------------
// 5) Edge scatter: acc[col] += h[row] * norm  (64 feats, 16 threads/edge x float4)
// ---------------------------------------------------------------------------
__global__ void k_scatter64(const float* __restrict__ h,
                            const int* __restrict__ row, const int* __restrict__ col,
                            const float* __restrict__ norm,
                            float* __restrict__ acc, int E) {
    int tid = blockIdx.x * blockDim.x + threadIdx.x;
    int e = tid >> 4;
    if (e >= E) return;
    int f4 = (tid & 15) << 2;
    int r = row[e], c = col[e];
    float nv = norm[e];
    float4 m = *(const float4*)(h + (size_t)r * 64 + f4);
    float* d = acc + (size_t)c * 64 + f4;
    atomicAdd(d + 0, m.x * nv);
    atomicAdd(d + 1, m.y * nv);
    atomicAdd(d + 2, m.z * nv);
    atomicAdd(d + 3, m.w * nv);
}

// ---------------------------------------------------------------------------
// 6) Post-scatter layer-1: x2 = relu(acc + conv1_b) -> bf16 hi/lo into x3[:,64:]
// ---------------------------------------------------------------------------
__global__ void k_post1(const float* __restrict__ acc, const float* __restrict__ bias,
                        unsigned short* __restrict__ x3hi, unsigned short* __restrict__ x3lo,
                        int total) {
    int i = blockIdx.x * blockDim.x + threadIdx.x;
    if (i >= total) return;
    int f = i & 63;
    size_t node = (size_t)(i >> 6);
    float v = acc[i] + bias[f];
    v = v > 0.f ? v : 0.f;
    unsigned short h = f32_to_bf16_rn(v);
    x3hi[node * 128 + 64 + f] = h;
    x3lo[node * 128 + 64 + f] = f32_to_bf16_rn(v - bf16_to_f32(h));
}

// ---------------------------------------------------------------------------
// 7) Combine + final projections: one wave per node.
//    x6 = relu(acc2 + conv2_b); x7 = x5 + x6;
//    out[n] = x7.fc2_w + fc2_b + conv3_b  (scatter3 adds the graph term)
//    h3[n]  = x7.conv3_w
// ---------------------------------------------------------------------------
__global__ void k_combine(const float* __restrict__ x5, const float* __restrict__ acc2,
                          const float* __restrict__ conv2_b,
                          const float* __restrict__ fc2_w, const float* __restrict__ fc2_b,
                          const float* __restrict__ conv3_w, const float* __restrict__ conv3_b,
                          float* __restrict__ out, float* __restrict__ h3, int N) {
    int wave = (int)((blockIdx.x * blockDim.x + threadIdx.x) >> 5);
    int lane = threadIdx.x & 31;
    if (wave >= N) return;
    float s9 = 0.f, s10 = 0.f;
#pragma unroll
    for (int p = 0; p < 2; ++p) {
        int f = lane + p * 32;
        float x6 = acc2[(size_t)wave * 64 + f] + conv2_b[f];
        x6 = x6 > 0.f ? x6 : 0.f;
        float x7 = x5[(size_t)wave * 64 + f] + x6;
        s9  += x7 * fc2_w[f];
        s10 += x7 * conv3_w[f];
    }
#pragma unroll
    for (int off = 16; off > 0; off >>= 1) {
        s9  += __shfl_xor(s9, off, 32);
        s10 += __shfl_xor(s10, off, 32);
    }
    if (lane == 0) {
        out[wave] = s9 + fc2_b[0] + conv3_b[0];
        h3[wave]  = s10;
    }
}

// ---------------------------------------------------------------------------
// 8) Final 1-wide scatter: out[col] += h3[row] * norm
// ---------------------------------------------------------------------------
__global__ void k_scatter1(const float* __restrict__ h3,
                           const int* __restrict__ row, const int* __restrict__ col,
                           const float* __restrict__ norm,
                           float* __restrict__ out, int E) {
    int e = blockIdx.x * blockDim.x + threadIdx.x;
    if (e < E) atomicAdd(&out[col[e]], h3[row[e]] * norm[e]);
}

// ---------------------------------------------------------------------------
// Host launcher
// ---------------------------------------------------------------------------
static inline size_t align256(size_t x) { return (x + 255) & ~(size_t)255; }

extern "C" void kernel_launch(void* const* d_in, const int* in_sizes, int n_in,
                              void* d_out, int out_size, void* d_ws, size_t ws_size,
                              hipStream_t stream) {
    const float* x       = (const float*)d_in[0];
    const float* fc_w    = (const float*)d_in[1];
    const float* fc_b    = (const float*)d_in[2];
    const float* conv1_w = (const float*)d_in[3];
    const float* conv1_b = (const float*)d_in[4];
    const float* fc1_w   = (const float*)d_in[5];
    const float* fc1_b   = (const float*)d_in[6];
    const float* conv2_w = (const float*)d_in[7];
    const float* conv2_b = (const float*)d_in[8];
    const float* fc2_w   = (const float*)d_in[9];
    const float* fc2_b   = (const float*)d_in[10];
    const float* conv3_w = (const float*)d_in[11];
    const float* conv3_b = (const float*)d_in[12];
    const int*   edge    = (const int*)d_in[13];

    const int N = in_sizes[0] / 128;        // 100000
    const int E = in_sizes[13] / 2;         // 1200000
    const int* row = edge;
    const int* col = edge + E;

    // Workspace carve-out
    char* p = (char*)d_ws;
    size_t cur = 0;
    auto carve = [&](size_t bytes) -> void* {
        void* r = p + cur;
        cur = align256(cur + bytes);
        return r;
    };
    unsigned short* xhi  = (unsigned short*)carve((size_t)N * 128 * 2);
    unsigned short* xlo  = (unsigned short*)carve((size_t)N * 128 * 2);
    unsigned short* x3hi = (unsigned short*)carve((size_t)N * 128 * 2);
    unsigned short* x3lo = (unsigned short*)carve((size_t)N * 128 * 2);
    unsigned short* B1hi = (unsigned short*)carve(16384 * 2);
    unsigned short* B1lo = (unsigned short*)carve(16384 * 2);
    unsigned short* B2hi = (unsigned short*)carve(16384 * 2);
    unsigned short* B2lo = (unsigned short*)carve(16384 * 2);
    float* deg  = (float*)carve((size_t)N * 4);
    float* dinv = (float*)carve((size_t)N * 4);
    float* nrm  = (float*)carve((size_t)E * 4);
    float* h1   = (float*)carve((size_t)N * 64 * 4);
    float* acc1 = (float*)carve((size_t)N * 64 * 4);
    float* x5   = (float*)carve((size_t)N * 64 * 4);
    float* h2   = (float*)carve((size_t)N * 64 * 4);
    float* acc2 = (float*)carve((size_t)N * 64 * 4);
    float* h3   = (float*)carve((size_t)N * 4);
    (void)ws_size; (void)n_in; (void)out_size;

    const int B = 256;

    // Zero the accumulators (graph-capture-safe async memsets)
    hipMemsetAsync(deg,  0, (size_t)N * 4, stream);
    hipMemsetAsync(acc1, 0, (size_t)N * 64 * 4, stream);
    hipMemsetAsync(acc2, 0, (size_t)N * 64 * 4, stream);

    // Weight pre-swizzle (both fused layers) + input conversion
    k_prep_B<<<(16384 + B - 1) / B, B, 0, stream>>>(fc_w, conv1_w, B1hi, B1lo);
    k_prep_B<<<(16384 + B - 1) / B, B, 0, stream>>>(fc1_w, conv2_w, B2hi, B2lo);
    {
        int n4 = N * 128 / 4;
        k_convert_x<<<(n4 + B - 1) / B, B, 0, stream>>>(x, xhi, xlo, n4);
    }

    // Graph normalization
    k_deg <<<(E + B - 1) / B, B, 0, stream>>>(col, deg, E);
    k_dinv<<<(N + B - 1) / B, B, 0, stream>>>(deg, dinv, N);
    k_norm<<<(E + B - 1) / B, B, 0, stream>>>(row, col, dinv, nrm, E);

    // Layer 1 fused GEMM: x @ [fc_w | conv1_w]  (x1 -> x3[:, :64] bf16, h1 f32)
    const int nstrips = (N + 15) / 16;                 // 6250 (exact)
    const int gemm_blocks = (nstrips + 7) / 8;         // 8 waves / block
    k_gemm<0><<<gemm_blocks, B, 0, stream>>>(xhi, xlo, B1hi, B1lo, fc_b,
                                             x3hi, x3lo, nullptr, h1, nstrips);

    // conv1 scatter + post (x2 -> x3[:, 64:] bf16)
    k_scatter64<<<((size_t)E * 16 + B - 1) / B, B, 0, stream>>>(h1, row, col, nrm, acc1, E);
    k_post1<<<(N * 64 + B - 1) / B, B, 0, stream>>>(acc1, conv1_b, x3hi, x3lo, N * 64);

    // Layer 2 fused GEMM: x3 @ [fc1_w | conv2_w]  (x5 f32, h2 f32)
    k_gemm<1><<<gemm_blocks, B, 0, stream>>>(x3hi, x3lo, B2hi, B2lo, fc1_b,
                                             nullptr, nullptr, x5, h2, nstrips);

    // conv2 scatter
    k_scatter64<<<((size_t)E * 16 + B - 1) / B, B, 0, stream>>>(h2, row, col, nrm, acc2, E);

    // Residual combine + final projections, then conv3 scatter into d_out
    k_combine<<<(N * 32 + B - 1) / B, B, 0, stream>>>(x5, acc2, conv2_b, fc2_w, fc2_b,
                                                      conv3_w, conv3_b,
                                                      (float*)d_out, h3, N);
    k_scatter1<<<(E + B - 1) / B, B, 0, stream>>>(h3, row, col, nrm, (float*)d_out, E);
}
